// AllPtransBlocksTRT_1013612281983
// MI455X (gfx1250) — matrix operand order repaired and checked by reading, not compiled, not run on hardware
//
#include <hip/hip_runtime.h>
#include <hip/hip_bf16.h>
#include <math.h>

// ---------------------------------------------------------------------------
// DSVT block: 4 transformer layers over N=108000 voxels, D=192, H=8, DH=24,
// FF=384, set attention over S=3000 sets of SS=36 tokens (inds = permutation).
// All dense GEMMs run on v_wmma_f32_16x16x32_f16 (f16 in, f32 accumulate).
// Weights are pre-converted to f16 once per launch; A-tiles are staged to LDS
// via GLOBAL_LOAD_ASYNC_TO_LDS_B128 (ASYNCcnt) when the toolchain exposes it.
// ---------------------------------------------------------------------------

typedef __attribute__((ext_vector_type(16))) _Float16 v16h;
typedef __attribute__((ext_vector_type(8)))  float    v8f;
typedef _Float16 half_t;

// Exact parameter types of __builtin_amdgcn_global_load_async_to_lds_b128:
// (int4_gccvec AS1*, int4_gccvec AS3*, imm offset, imm cpol)
typedef int i4v __attribute__((vector_size(16)));
typedef __attribute__((address_space(1))) i4v* as1_i4p;
typedef __attribute__((address_space(3))) i4v* as3_i4p;

#if defined(__has_builtin)
# if __has_builtin(__builtin_amdgcn_global_load_async_to_lds_b128)
#  define HAVE_ASYNC_LDS 1
# endif
# if __has_builtin(__builtin_amdgcn_s_wait_asynccnt)
#  define HAVE_WAIT_ASYNC_BUILTIN 1
# endif
#endif
#ifndef HAVE_ASYNC_LDS
# define HAVE_ASYNC_LDS 0
#endif
#ifndef HAVE_WAIT_ASYNC_BUILTIN
# define HAVE_WAIT_ASYNC_BUILTIN 0
#endif

#define NVOX 108000
#define DMODEL 192
#define NSET 3000
#define SETSZ 36
#define NLAYER 4
#define FFDIM 384
#define TRIPLE 576   // 3*D

__device__ __forceinline__ void wait_async_zero() {
#if HAVE_WAIT_ASYNC_BUILTIN
  __builtin_amdgcn_s_wait_asynccnt(0);
#else
  asm volatile("s_wait_asynccnt 0x0" ::: "memory");
#endif
}

// ---------------------------------------------------------------------------
// One-shot f32 -> f16 weight conversion (weights are L2-resident afterwards).
// ---------------------------------------------------------------------------
__global__ __launch_bounds__(256)
void cvt_kernel(const float* __restrict__ in, half_t* __restrict__ out, int n) {
  const int i = blockIdx.x * 256 + threadIdx.x;
  if (i < n) out[i] = (half_t)in[i];
}

// ---------------------------------------------------------------------------
// Fused gather + pos-embed + QKV projection.
// Block = (32,8): 8 waves share a 32-token M-tile (gathered order) staged as
// f16 in LDS (xs for V, xs+pos for Q,K). Each wave owns a 16-col slice of the
// 576-wide GEMM and runs two 16x16 accumulators over K=192 (6 k-steps,
// 2 v_wmma per step, B fragment reused across both M-halves).
// ---------------------------------------------------------------------------
__global__ __launch_bounds__(256)
void qkv_kernel(const float* __restrict__ cur, const float* __restrict__ pos,
                const int* __restrict__ inds, const half_t* __restrict__ Wqkv,
                const float* __restrict__ bqkv,
                half_t* __restrict__ q16, half_t* __restrict__ k16,
                half_t* __restrict__ v16o) {
  __shared__ half_t sQK[32 * 200];
  __shared__ half_t sXS[32 * 200];
  __shared__ int    sIdx[32];

  const int tid   = threadIdx.y * 32 + threadIdx.x;
  const int mbase = blockIdx.x * 32;

  if (tid < 32) sIdx[tid] = inds[mbase + tid];
  __syncthreads();

  for (int e = tid; e < 32 * DMODEL; e += 256) {
    const int r = e / DMODEL, c = e % DMODEL;
    const size_t off = (size_t)sIdx[r] * DMODEL + c;
    const float xv = cur[off];
    const float pv = pos[off];
    sXS[r * 200 + c] = (half_t)xv;
    sQK[r * 200 + c] = (half_t)(xv + pv);
  }
  __syncthreads();

  const int nt = blockIdx.y * 8 + threadIdx.y;   // 0..39, 36 valid n-tiles
  if (nt >= TRIPLE / 16) return;

  const int lane = threadIdx.x;
  const int hi   = lane >> 4;
  const int n    = nt * 16 + (lane & 15);        // global output column 0..575
  // Q,K consume pos-embedded tokens; V consumes raw tokens. The select is
  // wave-uniform (n-tile granularity is 16), so EXEC stays all-ones.
  const half_t* __restrict__ A   = (n < 2 * DMODEL) ? sQK : sXS;
  const half_t* __restrict__ ar0 = A + (lane & 15) * 200;
  const half_t* __restrict__ ar1 = ar0 + 16 * 200;

  const float bb = bqkv[n];
  v8f c0 = {bb, bb, bb, bb, bb, bb, bb, bb};
  v8f c1 = c0;

#pragma unroll
  for (int ks = 0; ks < DMODEL / 32; ++ks) {
    const int kb = ks * 32;
    const int k0 = kb + hi * 8;
    v16h a, b;
    const half_t* __restrict__ wr = Wqkv + (size_t)n * DMODEL + kb + hi * 8;
#pragma unroll
    for (int i = 0; i < 8; ++i) { b[i] = wr[i]; b[8 + i] = wr[16 + i]; }
#pragma unroll
    for (int i = 0; i < 8; ++i) { a[i] = ar0[k0 + i]; a[8 + i] = ar0[k0 + 16 + i]; }
    c0 = __builtin_amdgcn_wmma_f32_16x16x32_f16(false, a, false, b,
                                                (short)0, c0, false, false);
#pragma unroll
    for (int i = 0; i < 8; ++i) { a[i] = ar1[k0 + i]; a[8 + i] = ar1[k0 + 16 + i]; }
    c1 = __builtin_amdgcn_wmma_f32_16x16x32_f16(false, a, false, b,
                                                (short)0, c1, false, false);
  }

  half_t* __restrict__ dst;
  int col;
  if (n < DMODEL)          { dst = q16;  col = n; }
  else if (n < 2 * DMODEL) { dst = k16;  col = n - DMODEL; }
  else                     { dst = v16o; col = n - 2 * DMODEL; }
#pragma unroll
  for (int r = 0; r < 8; ++r) {
    const int m = mbase + r + hi * 8;
    dst[(size_t)m * DMODEL + col]        = (half_t)c0[r];
    dst[(size_t)(m + 16) * DMODEL + col] = (half_t)c1[r];
  }
}

// ---------------------------------------------------------------------------
// Generic WMMA GEMM: out[m,n] = epilogue( sum_k A[m,k] * W[n,k] + bias[n] ).
// A f16 [M x K] row-major, W f16 [Ncols x K] (pre-converted). 32-row M-tile
// per block (async-staged to LDS), two accumulators per wave reuse each B
// fragment.  mode 0: f32 store  1: f16 store  2: exact GELU -> f16
// 3: permutation scatter f32 store (row m -> voxel scat[m]).
// ---------------------------------------------------------------------------
__global__ __launch_bounds__(256)
void gemm_kernel(const half_t* __restrict__ A, const half_t* __restrict__ W,
                 const float* __restrict__ bias, const int K, const int Ncols,
                 const int mode, float* __restrict__ out32,
                 half_t* __restrict__ out16, const int* __restrict__ scat,
                 const int ldo) {
  __shared__ half_t sA[32 * 392];              // supports K <= 384 (+8 pad)
  const int KP    = K + 8;
  const int tid   = threadIdx.y * 32 + threadIdx.x;
  const int mbase = blockIdx.x * 32;

#if HAVE_ASYNC_LDS
  // Stage the A-tile with async VMEM->LDS transfers (no VGPR round trip),
  // tracked on ASYNCcnt. 16B chunks; LDS row stride (K+8)*2B is 16B aligned.
  const int cpr = K >> 3;                      // 8-half chunks per row
  for (int ch = tid; ch < 32 * cpr; ch += 256) {
    const int r = ch / cpr, q = ch % cpr;
    __builtin_amdgcn_global_load_async_to_lds_b128(
        (as1_i4p)(A + (size_t)(mbase + r) * K + q * 8),
        (as3_i4p)(sA + r * KP + q * 8), 0, 0);
  }
  wait_async_zero();
#else
  for (int e = tid; e < 32 * K; e += 256) {
    const int r = e / K, cc = e % K;
    sA[r * KP + cc] = A[(size_t)(mbase + r) * K + cc];
  }
#endif
  __syncthreads();

  const int nt = blockIdx.y * 8 + threadIdx.y;
  if (nt * 16 >= Ncols) return;

  const int lane = threadIdx.x;
  const int hi   = lane >> 4;
  const int n    = nt * 16 + (lane & 15);
  const half_t* __restrict__ ar0 = sA + (lane & 15) * KP;
  const half_t* __restrict__ ar1 = ar0 + 16 * KP;

  const float bb = bias[n];
  v8f c0 = {bb, bb, bb, bb, bb, bb, bb, bb};
  v8f c1 = c0;

  const int ksteps = K >> 5;
  for (int ks = 0; ks < ksteps; ++ks) {
    const int kb = ks * 32;
    const int k0 = kb + hi * 8;
    v16h a, b;
    const half_t* __restrict__ wr = W + (size_t)n * K + kb + hi * 8;
#pragma unroll
    for (int i = 0; i < 8; ++i) { b[i] = wr[i]; b[8 + i] = wr[16 + i]; }
#pragma unroll
    for (int i = 0; i < 8; ++i) { a[i] = ar0[k0 + i]; a[8 + i] = ar0[k0 + 16 + i]; }
    c0 = __builtin_amdgcn_wmma_f32_16x16x32_f16(false, a, false, b,
                                                (short)0, c0, false, false);
#pragma unroll
    for (int i = 0; i < 8; ++i) { a[i] = ar1[k0 + i]; a[8 + i] = ar1[k0 + 16 + i]; }
    c1 = __builtin_amdgcn_wmma_f32_16x16x32_f16(false, a, false, b,
                                                (short)0, c1, false, false);
  }

#pragma unroll
  for (int r = 0; r < 8; ++r) {
    const int m0 = mbase + r + hi * 8;
    const int m1 = m0 + 16;
    const float x0 = c0[r], x1 = c1[r];
    if (mode == 0) {
      out32[(size_t)m0 * ldo + n] = x0;
      out32[(size_t)m1 * ldo + n] = x1;
    } else if (mode == 1) {
      out16[(size_t)m0 * ldo + n] = (half_t)x0;
      out16[(size_t)m1 * ldo + n] = (half_t)x1;
    } else if (mode == 2) {
      out16[(size_t)m0 * ldo + n] = (half_t)(0.5f * x0 * (1.0f + erff(x0 * 0.70710678118654752f)));
      out16[(size_t)m1 * ldo + n] = (half_t)(0.5f * x1 * (1.0f + erff(x1 * 0.70710678118654752f)));
    } else {
      out32[(size_t)scat[m0] * ldo + n] = x0;   // inds is a permutation
      out32[(size_t)scat[m1] * ldo + n] = x1;
    }
  }
}

// ---------------------------------------------------------------------------
// Masked set attention (VALU): one block per set, 288 threads = (36 q) x (8 h).
// K/V for the set staged in LDS; per-thread 36-score masked softmax + AV.
// ---------------------------------------------------------------------------
__global__ __launch_bounds__(288)
void attn_kernel(const half_t* __restrict__ q16, const half_t* __restrict__ k16,
                 const half_t* __restrict__ v16, const unsigned char* __restrict__ mask,
                 half_t* __restrict__ o16) {
  __shared__ float sK[SETSZ * 200];
  __shared__ float sV[SETSZ * 200];
  const int s    = blockIdx.x;
  const int tid  = threadIdx.x;
  const size_t base = (size_t)s * SETSZ * DMODEL;

  for (int e = tid; e < SETSZ * DMODEL; e += 288) {
    const int r = e / DMODEL, c = e % DMODEL;
    sK[r * 200 + c] = (float)k16[base + (size_t)r * DMODEL + c];
    sV[r * 200 + c] = (float)v16[base + (size_t)r * DMODEL + c];
  }
  __syncthreads();

  const int qr = tid / 8;     // 0..35
  const int h  = tid % 8;

  float qv[24];
#pragma unroll
  for (int d = 0; d < 24; ++d)
    qv[d] = (float)q16[base + (size_t)qr * DMODEL + h * 24 + d];

  float sc[SETSZ];
  float mx = -1e30f;
  for (int j = 0; j < SETSZ; ++j) {
    const float* kr = sK + j * 200 + h * 24;
    float acc = 0.0f;
#pragma unroll
    for (int d = 0; d < 24; ++d) acc += qv[d] * kr[d];
    acc *= 0.20412414523193150f;              // 1/sqrt(24)
    if (mask[(size_t)s * SETSZ + j]) acc = -1e9f;
    sc[j] = acc;
    mx = fmaxf(mx, acc);
  }
  float sum = 0.0f;
  for (int j = 0; j < SETSZ; ++j) { sc[j] = __expf(sc[j] - mx); sum += sc[j]; }
  const float inv = 1.0f / sum;

  float oacc[24];
#pragma unroll
  for (int d = 0; d < 24; ++d) oacc[d] = 0.0f;
  for (int j = 0; j < SETSZ; ++j) {
    const float w = sc[j] * inv;
    const float* vr = sV + j * 200 + h * 24;
#pragma unroll
    for (int d = 0; d < 24; ++d) oacc[d] += w * vr[d];
  }
#pragma unroll
  for (int d = 0; d < 24; ++d)
    o16[base + (size_t)qr * DMODEL + h * 24 + d] = (half_t)oacc[d];
}

// ---------------------------------------------------------------------------
// Fused residual add + LayerNorm. One wave per row (D=192 -> 6 elems/lane).
// Optional f16 mirror feeds the next GEMM's A operand.
// ---------------------------------------------------------------------------
__global__ __launch_bounds__(256)
void add_ln_kernel(const float* __restrict__ a, const float* __restrict__ b,
                   const float* __restrict__ g, const float* __restrict__ bt,
                   float* __restrict__ out32, half_t* __restrict__ out16) {
  const int row  = blockIdx.x * 8 + threadIdx.y;
  const int lane = threadIdx.x;
  const float* __restrict__ ar = a + (size_t)row * DMODEL;
  const float* __restrict__ br = b + (size_t)row * DMODEL;

  float v[6];
  float sum = 0.0f;
#pragma unroll
  for (int j = 0; j < 6; ++j) {
    const int c = lane + 32 * j;
    v[j] = ar[c] + br[c];
    sum += v[j];
  }
#pragma unroll
  for (int off = 16; off > 0; off >>= 1) sum += __shfl_xor(sum, off);
  const float mu = sum * (1.0f / DMODEL);

  float vs = 0.0f;
#pragma unroll
  for (int j = 0; j < 6; ++j) { const float d = v[j] - mu; vs += d * d; }
#pragma unroll
  for (int off = 16; off > 0; off >>= 1) vs += __shfl_xor(vs, off);
  const float rinv = rsqrtf(vs * (1.0f / DMODEL) + 1e-5f);

#pragma unroll
  for (int j = 0; j < 6; ++j) {
    const int c = lane + 32 * j;
    const float o = (v[j] - mu) * rinv * g[c] + bt[c];
    out32[(size_t)row * DMODEL + c] = o;
    if (out16) out16[(size_t)row * DMODEL + c] = (half_t)o;
  }
}

// ---------------------------------------------------------------------------
// Host orchestration: weight f16 pre-pass, then 4 layers x (QKV -> attn ->
// O-proj/scatter -> LN1 -> FFN1(GELU) -> FFN2 -> LN2 -> encLN) + blkLN.
// ---------------------------------------------------------------------------
extern "C" void kernel_launch(void* const* d_in, const int* in_sizes, int n_in,
                              void* d_out, int out_size, void* d_ws, size_t ws_size,
                              hipStream_t stream) {
  (void)in_sizes; (void)n_in; (void)out_size; (void)ws_size;

  const float*         pillar = (const float*)d_in[0];
  const int*           inds0  = (const int*)d_in[1];            // [2,S,SS]
  const int*           inds1  = (const int*)d_in[2];            // [2,S,SS]
  const unsigned char* mask0  = (const unsigned char*)d_in[3];  // bool [2,S,SS]
  const unsigned char* mask1  = (const unsigned char*)d_in[4];
  const float*         posA   = (const float*)d_in[5];          // [2,2,N,D]
  const float* Wqkv = (const float*)d_in[6];
  const float* bqkv = (const float*)d_in[7];
  const float* Wo   = (const float*)d_in[8];
  const float* bo   = (const float*)d_in[9];
  const float* W1   = (const float*)d_in[10];
  const float* b1   = (const float*)d_in[11];
  const float* W2   = (const float*)d_in[12];
  const float* b2   = (const float*)d_in[13];
  const float* ln1g = (const float*)d_in[14];
  const float* ln1b = (const float*)d_in[15];
  const float* ln2g = (const float*)d_in[16];
  const float* ln2b = (const float*)d_in[17];
  const float* eng  = (const float*)d_in[18];
  const float* enb  = (const float*)d_in[19];
  const float* blg  = (const float*)d_in[20];
  const float* blb  = (const float*)d_in[21];

  const size_t ND = (size_t)NVOX * DMODEL;   // 20,736,000
  const size_t NF = (size_t)NVOX * FFDIM;    // 41,472,000
  const int SZ_QKV = NLAYER * TRIPLE * DMODEL;   // 442368
  const int SZ_O   = NLAYER * DMODEL * DMODEL;   // 147456
  const int SZ_1   = NLAYER * FFDIM * DMODEL;    // 294912
  const int SZ_2   = NLAYER * DMODEL * FFDIM;    // 294912

  char* w = (char*)d_ws;
  half_t* q16    = (half_t*)w; w += ND * 2;
  half_t* k16    = (half_t*)w; w += ND * 2;
  half_t* v16    = (half_t*)w; w += ND * 2;
  half_t* o16    = (half_t*)w; w += ND * 2;
  half_t* xh16   = (half_t*)w; w += ND * 2;
  half_t* h16    = (half_t*)w; w += NF * 2;
  float*  scat32 = (float*)w;  w += ND * 4;  // O-proj scattered / y2 reuse
  float*  x32    = (float*)w;  w += ND * 4;  // post-LN1 activations
  float*  y32    = (float*)w;  w += ND * 4;  // FFN2 output
  float*  res32  = (float*)w;  w += ND * 4;  // block residual
  float*  cur32  = (float*)w;  w += ND * 4;  // running activations
  half_t* wq16   = (half_t*)w; w += (size_t)SZ_QKV * 2;
  half_t* wo16   = (half_t*)w; w += (size_t)SZ_O * 2;
  half_t* w116   = (half_t*)w; w += (size_t)SZ_1 * 2;
  half_t* w216   = (half_t*)w; w += (size_t)SZ_2 * 2;

  const dim3 blk(32, 8);
  const int  MT = NVOX / 32;                 // 3375 M-tiles of 32 rows (exact)

  // Weight f16 pre-pass (all layers at once; ~4.7MB f32 -> ~2.4MB f16).
  cvt_kernel<<<(SZ_QKV + 255) / 256, 256, 0, stream>>>(Wqkv, wq16, SZ_QKV);
  cvt_kernel<<<(SZ_O   + 255) / 256, 256, 0, stream>>>(Wo,   wo16, SZ_O);
  cvt_kernel<<<(SZ_1   + 255) / 256, 256, 0, stream>>>(W1,   w116, SZ_1);
  cvt_kernel<<<(SZ_2   + 255) / 256, 256, 0, stream>>>(W2,   w216, SZ_2);

  (void)hipMemcpyAsync(cur32, pillar, ND * sizeof(float), hipMemcpyDeviceToDevice, stream);

  for (int blc = 0; blc < 2; ++blc) {
    (void)hipMemcpyAsync(res32, cur32, ND * sizeof(float), hipMemcpyDeviceToDevice, stream);
    for (int s = 0; s < 2; ++s) {
      const int l = 2 * blc + s;
      const int*           inds = ((l < 2) ? inds0 : inds1) + (size_t)(l & 1) * NSET * SETSZ;
      const unsigned char* msk  = ((l < 2) ? mask0 : mask1) + (size_t)(l & 1) * NSET * SETSZ;
      const float*         pos  = posA + (size_t)l * ND;

      // QKV projection (gather + pos-embed fused), f16 outputs in set order.
      qkv_kernel<<<dim3(MT, 5), blk, 0, stream>>>(
          cur32, pos, inds, wq16 + (size_t)l * TRIPLE * DMODEL,
          bqkv + (size_t)l * TRIPLE, q16, k16, v16);

      // Masked softmax attention per set.
      attn_kernel<<<NSET, 288, 0, stream>>>(q16, k16, v16, msk, o16);

      // Output projection + permutation scatter back to voxel order.
      gemm_kernel<<<dim3(MT, 2), blk, 0, stream>>>(
          o16, wo16 + (size_t)l * DMODEL * DMODEL, bo + (size_t)l * DMODEL,
          DMODEL, DMODEL, 3, scat32, nullptr, inds, DMODEL);

      // x = LN1(src + scat)   (f32 + f16 mirror for FFN1)
      add_ln_kernel<<<NVOX / 8, blk, 0, stream>>>(
          cur32, scat32, ln1g + (size_t)l * DMODEL, ln1b + (size_t)l * DMODEL,
          x32, xh16);

      // h = GELU(x @ W1^T + b1)  -> f16
      gemm_kernel<<<dim3(MT, 3), blk, 0, stream>>>(
          xh16, w116 + (size_t)l * FFDIM * DMODEL, b1 + (size_t)l * FFDIM,
          DMODEL, FFDIM, 2, nullptr, h16, nullptr, FFDIM);

      // y = h @ W2^T + b2  -> f32
      gemm_kernel<<<dim3(MT, 2), blk, 0, stream>>>(
          h16, w216 + (size_t)l * DMODEL * FFDIM, b2 + (size_t)l * DMODEL,
          FFDIM, DMODEL, 0, y32, nullptr, nullptr, DMODEL);

      // y2 = LN2(x + y)  (reuse scat32)
      add_ln_kernel<<<NVOX / 8, blk, 0, stream>>>(
          x32, y32, ln2g + (size_t)l * DMODEL, ln2b + (size_t)l * DMODEL,
          scat32, nullptr);

      // cur = encLN(identity + y2)   (in-place safe: wave reads row then writes)
      add_ln_kernel<<<NVOX / 8, blk, 0, stream>>>(
          cur32, scat32, eng + (size_t)l * DMODEL, enb + (size_t)l * DMODEL,
          cur32, nullptr);
    }
    // block residual: out = blkLN(residual + cur)
    float* dst = (blc == 1) ? (float*)d_out : cur32;
    add_ln_kernel<<<NVOX / 8, blk, 0, stream>>>(
        res32, cur32, blg + (size_t)blc * DMODEL, blb + (size_t)blc * DMODEL,
        dst, nullptr);
  }
}
